// ProgAdd2x2Model_81990925680945
// MI455X (gfx1250) — matrix operand order repaired and checked by reading, not compiled, not run on hardware
//
#include <hip/hip_runtime.h>
#include <hip/hip_bf16.h>

typedef _Float16 half8  __attribute__((ext_vector_type(8)));
typedef _Float16 half16 __attribute__((ext_vector_type(16)));
typedef float    v8f    __attribute__((ext_vector_type(8)));

union Frag {
    half16 v;
    half8  h[2];
};

// ---------------------------------------------------------------------------
// Prep: repack W1 (784x256 fp32, row-major) into f16 WMMA B-fragment layout,
// K padded 784 -> 800 (25 k-chunks of 32). Layout per (kt, nt) 32x16 tile:
//   lane 0-15 : col n = nt*16+lane,    K = kt*32 + {0..15}
//   lane 16-31: col n = nt*16+lane-16, K = kt*32 + {16..31}
// 16 consecutive halves per lane => each wave's tile load is 1KB coalesced.
// Flat index: (((kt*16 + nt)*32 + lane)*16 + e
// ---------------------------------------------------------------------------
__global__ __launch_bounds__(256) void prep_w1_kernel(
    const float* __restrict__ W1, _Float16* __restrict__ W1p)
{
    int idx = blockIdx.x * 256 + threadIdx.x;      // 0 .. 204799
    if (idx >= 25 * 16 * 32 * 16) return;
    int e    = idx & 15;
    int lane = (idx >> 4) & 31;
    int nt   = (idx >> 9) & 15;
    int kt   = idx >> 13;
    int k = kt * 32 + ((lane & 16) ? 16 : 0) + e;
    int n = nt * 16 + (lane & 15);
    float v = (k < 784) ? W1[k * 256 + n] : 0.0f;
    W1p[idx] = (_Float16)v;
}

// ---------------------------------------------------------------------------
// Fused main kernel: 32 rows of x per block, 8 waves.
//   Phase 1: stage x tile (32x784 fp32 -> f16, K padded to 800) in LDS
//   Phase 2: h1 = relu(x@W1 + b1) via v_wmma_f32_16x16x32_f16
//            wave w: rows slab*16..+15 (slab=w>>2), cols quad*64..+63 (quad=w&3)
//   Phase 3: bias + relu, h1 tile (32x256 f32) into LDS (reuses x region)
//   Phase 4: logits = h1@W2 + b2 (K=256 split over 8 threads/row), softmax/0.1
//   Phase 5: pointer net + dist conv + log, 8 batch items per block
// ---------------------------------------------------------------------------
#define SMEM_BYTES 54528   // max( 32*808*2 = 51712 , 32768+10240+1280+10240 )

__global__ __launch_bounds__(256) void fused_main_kernel(
    const float* __restrict__ grid,   // (B*4, 784)
    const float* __restrict__ prog,   // (B, 4)
    const float* __restrict__ b1,     // (256)
    const float* __restrict__ W2,     // (256,10)
    const float* __restrict__ b2,     // (10)
    const float* __restrict__ Wp,     // (4,16)
    const float* __restrict__ bp,     // (16)
    const float* __restrict__ Wr1,    // (16,4)
    const float* __restrict__ br1,    // (4)
    const float* __restrict__ Wr2,    // (16,4)
    const float* __restrict__ br2,    // (4)
    const _Float16* __restrict__ W1p, // repacked (see prep)
    float* __restrict__ out)          // (B,19)
{
    __shared__ __align__(16) unsigned char smem[SMEM_BYTES];
    _Float16* xs = (_Float16*)smem;                    // [32][808] f16
    float* h1    = (float*)smem;                       // [32][256] f32
    float* ps    = (float*)(smem + 32768);             // [32][8][10]
    float* dp    = (float*)(smem + 43008);             // [32][10]
    float* w2s   = (float*)(smem + 44288);             // [256][10]

    const int tid  = threadIdx.x;
    const int lane = tid & 31;
    const int wave = tid >> 5;
    const int slab = wave >> 2;    // 0..1 -> row half
    const int quad = wave & 3;     // 0..3 -> 64-col stripe

    // ---- Phase 1: x tile -> LDS f16 (784 = 196 float4 per row) ------------
    const float* xrow = grid + (size_t)blockIdx.x * 32u * 784u;
    for (int i = tid; i < 32 * 196; i += 256) {
        int row = i / 196;
        int c4  = i - row * 196;
        float4 v = *(const float4*)(xrow + row * 784 + c4 * 4);
        _Float16* d = xs + row * 808 + c4 * 4;
        d[0] = (_Float16)v.x; d[1] = (_Float16)v.y;
        d[2] = (_Float16)v.z; d[3] = (_Float16)v.w;
    }
    for (int i = tid; i < 32 * 24; i += 256) {         // zero K pad 784..807
        int row = i / 24;
        xs[row * 808 + 784 + (i - row * 24)] = (_Float16)0.0f;
    }
    __syncthreads();

    // ---- Phase 2: WMMA GEMM ----------------------------------------------
    v8f acc[4] = {};
    const int rowL = slab * 16 + (lane & 15);
    const int klo  = (lane & 16) ? 16 : 0;             // A layout: contiguous K per lane half
    const _Float16* xbase = xs + rowL * 808 + klo;

    for (int kt = 0; kt < 25; ++kt) {
        Frag a;
        const _Float16* ap = xbase + kt * 32;
        a.h[0] = *(const half8*)ap;                    // K {0..7}  (+klo)
        a.h[1] = *(const half8*)(ap + 8);              // K {8..15} (+klo)

        const _Float16* wb = W1p + (size_t)((kt * 16 + quad * 4) * 32 + lane) * 16;
        if (kt + 1 < 25)
            __builtin_prefetch(W1p + (size_t)(((kt + 1) * 16 + quad * 4) * 32 + lane) * 16, 0, 1);

#pragma unroll
        for (int t = 0; t < 4; ++t) {
            Frag b;
            const _Float16* bb = wb + t * 512;         // next 32x16 tile
            b.h[0] = *(const half8*)bb;
            b.h[1] = *(const half8*)(bb + 8);
            acc[t] = __builtin_amdgcn_wmma_f32_16x16x32_f16(
                false, a.v, false, b.v, (short)0, acc[t], false, false);
        }
    }
    __syncthreads();                                   // xs dead after this

    // ---- Phase 3: bias + relu -> LDS h1; stage W2 -> LDS ------------------
    {
        const int ncol  = quad * 64 + (lane & 15);
        const int rbase = slab * 16 + ((lane & 16) ? 8 : 0);
        float bias[4];
#pragma unroll
        for (int t = 0; t < 4; ++t) bias[t] = b1[ncol + t * 16];
#pragma unroll
        for (int t = 0; t < 4; ++t)
#pragma unroll
            for (int r = 0; r < 8; ++r) {
                float h = acc[t][r] + bias[t];
                h1[(rbase + r) * 256 + ncol + t * 16] = h > 0.0f ? h : 0.0f;
            }
        for (int i = tid; i < 2560; i += 256) w2s[i] = W2[i];
    }
    __syncthreads();

    // ---- Phase 4a: partial 256->10 GEMV (8 threads per row) ---------------
    {
        int row = tid >> 3, part = tid & 7;
        float pd[10];
#pragma unroll
        for (int d = 0; d < 10; ++d) pd[d] = 0.0f;
        for (int i = 0; i < 32; ++i) {
            int k = part + i * 8;
            float hv = h1[row * 256 + k];
#pragma unroll
            for (int d = 0; d < 10; ++d) pd[d] += hv * w2s[k * 10 + d];
        }
#pragma unroll
        for (int d = 0; d < 10; ++d) ps[(row * 8 + part) * 10 + d] = pd[d];
    }
    __syncthreads();

    // ---- Phase 4b: reduce + softmax(logits / 0.1) -------------------------
    if (tid < 32) {
        float lg[10];
#pragma unroll
        for (int d = 0; d < 10; ++d) lg[d] = b2[d];
        for (int p = 0; p < 8; ++p)
#pragma unroll
            for (int d = 0; d < 10; ++d) lg[d] += ps[(tid * 8 + p) * 10 + d];
        float m = lg[0];
#pragma unroll
        for (int d = 1; d < 10; ++d) m = fmaxf(m, lg[d]);
        float e[10], s = 0.0f;
#pragma unroll
        for (int d = 0; d < 10; ++d) { e[d] = __expf((lg[d] - m) * 10.0f); s += e[d]; }
        float inv = 1.0f / s;
#pragma unroll
        for (int d = 0; d < 10; ++d) dp[tid * 10 + d] = e[d] * inv;
    }
    __syncthreads();

    // ---- Phase 5: pointer net + conv + log, 8 batch items -----------------
    if (tid < 8) {
        int gb = blockIdx.x * 8 + tid;
        const float* oh = prog + gb * 4;
        float h[16];
#pragma unroll
        for (int j = 0; j < 16; ++j)
            h[j] = bp[j] + oh[0] * Wp[j] + oh[1] * Wp[16 + j]
                         + oh[2] * Wp[32 + j] + oh[3] * Wp[48 + j];
        float l1[4], l2[4];
#pragma unroll
        for (int p = 0; p < 4; ++p) { l1[p] = br1[p]; l2[p] = br2[p]; }
        for (int j = 0; j < 16; ++j)
#pragma unroll
            for (int p = 0; p < 4; ++p) {
                l1[p] += h[j] * Wr1[j * 4 + p];
                l2[p] += h[j] * Wr2[j * 4 + p];
            }
        float p1[4], p2[4];
        {
            float m1 = fmaxf(fmaxf(l1[0], l1[1]), fmaxf(l1[2], l1[3]));
            float m2 = fmaxf(fmaxf(l2[0], l2[1]), fmaxf(l2[2], l2[3]));
            float s1 = 0.0f, s2 = 0.0f;
#pragma unroll
            for (int p = 0; p < 4; ++p) {
                p1[p] = __expf(l1[p] - m1); s1 += p1[p];
                p2[p] = __expf(l2[p] - m2); s2 += p2[p];
            }
#pragma unroll
            for (int p = 0; p < 4; ++p) { p1[p] /= s1; p2[p] /= s2; }
        }
        float d1[10], d2[10];
#pragma unroll
        for (int d = 0; d < 10; ++d) {
            float a = 0.0f, b = 0.0f;
#pragma unroll
            for (int p = 0; p < 4; ++p) {
                float dv = dp[(tid * 4 + p) * 10 + d];
                a += p1[p] * dv;
                b += p2[p] * dv;
            }
            d1[d] = a; d2[d] = b;
        }
        float* o = out + (size_t)gb * 19;
#pragma unroll
        for (int k = 0; k < 19; ++k) {
            int lo = k > 9 ? k - 9 : 0;
            int hi = k < 9 ? k : 9;
            float s = 0.0f;
            for (int i = lo; i <= hi; ++i) s += d1[i] * d2[k - i];
            o[k] = __logf(s + 1e-10f);
        }
    }
}

extern "C" void kernel_launch(void* const* d_in, const int* in_sizes, int n_in,
                              void* d_out, int out_size, void* d_ws, size_t ws_size,
                              hipStream_t stream)
{
    const float* grid = (const float*)d_in[0];
    const float* prog = (const float*)d_in[1];
    const float* W1   = (const float*)d_in[2];
    const float* b1   = (const float*)d_in[3];
    const float* W2   = (const float*)d_in[4];
    const float* b2   = (const float*)d_in[5];
    const float* Wp   = (const float*)d_in[6];
    const float* bp   = (const float*)d_in[7];
    const float* Wr1  = (const float*)d_in[8];
    const float* br1  = (const float*)d_in[9];
    const float* Wr2  = (const float*)d_in[10];
    const float* br2  = (const float*)d_in[11];
    float* out = (float*)d_out;

    _Float16* W1p = (_Float16*)d_ws;        // 25*16*32*16 = 204800 halves (400KB)

    prep_w1_kernel<<<800, 256, 0, stream>>>(W1, W1p);

    int rows   = in_sizes[0] / 784;         // B*4 = 65536
    int blocks = rows / 32;                 // 2048
    fused_main_kernel<<<blocks, 256, 0, stream>>>(
        grid, prog, b1, W2, b2, Wp, bp, Wr1, br1, Wr2, br2, W1p, out);
}
